// Attention_43963285242640
// MI455X (gfx1250) — matrix operand order repaired and checked by reading, not compiled, not run on hardware
//
#include <hip/hip_runtime.h>

// ---------------------------------------------------------------------------
// Shapes (fixed by the reference)
// ---------------------------------------------------------------------------
#define B_   2
#define T_   2048
#define HID_ 4096
#define H_   32
#define KV_  4
#define D_   128
#define SLOTS_ (H_ + 2 * KV_)           // 40
#define QKVO_  (SLOTS_ * D_)            // 5120
#define M_     (B_ * T_)                // 4096
#define SCALE_ 0.08838834764831845f     // 1/sqrt(128)
#define LN10K_ 9.210340371976184f       // ln(10000)

typedef __attribute__((ext_vector_type(16))) _Float16 v16h;
typedef __attribute__((ext_vector_type(8)))  _Float16 v8h;
typedef __attribute__((ext_vector_type(4)))  _Float16 v4h;
typedef __attribute__((ext_vector_type(8)))  float    v8f;
typedef __attribute__((ext_vector_type(4)))  float    v4f;
typedef __attribute__((ext_vector_type(4)))  unsigned int v4u;
typedef __attribute__((ext_vector_type(8)))  int      v8i;
typedef __attribute__((ext_vector_type(4)))  int      v4i;

__device__ __forceinline__ v8f wmma_f16(v16h a, v16h b, v8f c) {
  return __builtin_amdgcn_wmma_f32_16x16x32_f16(false, a, false, b, (short)0, c,
                                                false, false);
}

// A-matrix fragment with contiguous K order (matches the B loads):
//   lanes 0-15 (lh=0): K = {0..15};  lanes 16-31 (lh=1): K = {16..31}
__device__ __forceinline__ v16h a_frag(const _Float16* row, int lh) {
  const v8h lo = *(const v8h*)(row + lh * 16);
  const v8h hi = *(const v8h*)(row + lh * 16 + 8);
  return __builtin_shufflevector(lo, hi, 0, 1, 2, 3, 4, 5, 6, 7, 8, 9, 10, 11,
                                 12, 13, 14, 15);
}

// ---------------------------------------------------------------------------
// Tensor Data Mover: copy a 128-row x 32-col f16 tile (row stride = K elems)
// from global memory into LDS.  clang-23 6-arg builtin
// (uint32x4 g0, int32x8 g1, int32x4, int32x4, int32x8, i32 cpol).
// D# per ISA ch.8:
//   group0 = {count=1, lds_addr, global_addr[56:0], type=2}
//   group1 = {data_size=1(2B), tensor_dim0=K, tensor_dim1=M, tile 32x128,
//             tensor_dim0_stride=K};  remaining groups zero (2-D tile).
// ---------------------------------------------------------------------------
__device__ __forceinline__ void tdm_load_tile_f16(unsigned int lds_off,
                                                  const _Float16* gsrc, int K,
                                                  int M) {
  const unsigned long long ga = (unsigned long long)(uintptr_t)gsrc;
  v4u g0;
  g0[0] = 1u;                                        // count = 1 valid D#
  g0[1] = lds_off;                                   // LDS byte address
  g0[2] = (unsigned int)ga;                          // global addr lo
  g0[3] = (unsigned int)(ga >> 32) | 0x80000000u;    // addr hi | type=2
  v8i g1;
  g1[0] = 0x00010000;                                         // data_size = 2B
  g1[1] = (int)(((unsigned)K & 0xffffu) << 16);               // dim0 lo16
  g1[2] = (int)((((unsigned)K >> 16) & 0xffffu) |
                (((unsigned)M & 0xffffu) << 16));             // dim0 hi | dim1 lo
  g1[3] = (int)((((unsigned)M >> 16) & 0xffffu) | (32u << 16)); // dim1 hi | tile0
  g1[4] = 128;                                                // tile_dim1 = 128
  g1[5] = K;                                                  // dim0_stride lo32
  g1[6] = 0;
  g1[7] = 0;
  v4i z4 = {0, 0, 0, 0};
  v8i z8 = {0, 0, 0, 0, 0, 0, 0, 0};
  __builtin_amdgcn_tensor_load_to_lds(g0, g1, z4, z4, z8, 0);
}

// ---------------------------------------------------------------------------
// Kernel 1 & 4: C[M,N] = A[M,K] * W[N,K]^T   (W fp32; A fp32 or f16)
// 128x128 block tile, 8 waves, wave tile 64x32 -> 4x2 WMMA accumulators.
// Register double-buffered staging; A_HALF path stages A via the TDM.
// ---------------------------------------------------------------------------
template <bool A_HALF>
__global__ __launch_bounds__(256) void gemm_kernel(const void* __restrict__ Ap,
                                                   const float* __restrict__ W,
                                                   float* __restrict__ C,
                                                   int M, int N, int K) {
  __shared__ __align__(64) _Float16 As[128 * 32];
  __shared__ __align__(64) _Float16 Bs[128 * 32];

  const int tid  = threadIdx.x;
  const int m0   = blockIdx.y * 128;
  const int n0   = blockIdx.x * 128;
  const int wave = tid >> 5;
  const int lane = tid & 31;
  const int wm   = (wave >> 2) * 64;   // 0 / 64
  const int wn   = (wave & 3) * 32;    // 0..96
  const int lr   = lane & 15;
  const int lh   = lane >> 4;

  const float*    Af = (const float*)Ap;
  const _Float16* Ah = (const _Float16*)Ap;

  // staging slots: 1024 float4/half4 positions per tile, 4 per thread
  int fr[4], fc[4];
#pragma unroll
  for (int i = 0; i < 4; ++i) {
    const int f = tid + i * 256;
    fr[i] = f >> 3;          // row in 128
    fc[i] = (f & 7) * 4;     // col in 32
  }

  v8f acc[4][2] = {};
  v4f aR[4], bR[4];

  // ---- prologue: fetch tile 0 --------------------------------------------
  if constexpr (A_HALF) {
    if (tid < 32)
      tdm_load_tile_f16((unsigned int)(uintptr_t)&As[0],
                        Ah + (size_t)m0 * K, K, M);
  } else {
#pragma unroll
    for (int i = 0; i < 4; ++i)
      aR[i] = *(const v4f*)&Af[(size_t)(m0 + fr[i]) * K + fc[i]];
  }
#pragma unroll
  for (int i = 0; i < 4; ++i)
    bR[i] = *(const v4f*)&W[(size_t)(n0 + fr[i]) * K + fc[i]];
  if constexpr (!A_HALF) {
#pragma unroll
    for (int i = 0; i < 4; ++i)
      *(v4h*)&As[fr[i] * 32 + fc[i]] = __builtin_convertvector(aR[i], v4h);
  }
#pragma unroll
  for (int i = 0; i < 4; ++i)
    *(v4h*)&Bs[fr[i] * 32 + fc[i]] = __builtin_convertvector(bR[i], v4h);
  if constexpr (A_HALF) {
    if (tid < 32) __builtin_amdgcn_s_wait_tensorcnt((short)0);
  }
  __syncthreads();

  // ---- main K loop --------------------------------------------------------
  for (int kk = 0; kk < K; kk += 32) {
    v16h a[4], b[2];
#pragma unroll
    for (int i = 0; i < 4; ++i) a[i] = a_frag(&As[(wm + i * 16 + lr) * 32], lh);
#pragma unroll
    for (int j = 0; j < 2; ++j)
      b[j] = *(const v16h*)&Bs[(wn + j * 16 + lr) * 32 + lh * 16];
#pragma unroll
    for (int i = 0; i < 4; ++i)
#pragma unroll
      for (int j = 0; j < 2; ++j) acc[i][j] = wmma_f16(a[i], b[j], acc[i][j]);

    const bool more = (kk + 32) < K;
    if (more) {  // prefetch next tile into registers (overlaps the WMMAs)
      if constexpr (!A_HALF) {
#pragma unroll
        for (int i = 0; i < 4; ++i)
          aR[i] = *(const v4f*)&Af[(size_t)(m0 + fr[i]) * K + kk + 32 + fc[i]];
      }
#pragma unroll
      for (int i = 0; i < 4; ++i)
        bR[i] = *(const v4f*)&W[(size_t)(n0 + fr[i]) * K + kk + 32 + fc[i]];
    }
    __syncthreads();
    if (more) {
      if constexpr (!A_HALF) {
#pragma unroll
        for (int i = 0; i < 4; ++i)
          *(v4h*)&As[fr[i] * 32 + fc[i]] = __builtin_convertvector(aR[i], v4h);
      }
#pragma unroll
      for (int i = 0; i < 4; ++i)
        *(v4h*)&Bs[fr[i] * 32 + fc[i]] = __builtin_convertvector(bR[i], v4h);
      if constexpr (A_HALF) {
        if (tid < 32) {
          tdm_load_tile_f16((unsigned int)(uintptr_t)&As[0],
                            Ah + (size_t)m0 * K + kk + 32, K, M);
          __builtin_amdgcn_s_wait_tensorcnt((short)0);
        }
      }
      __syncthreads();
    }
  }

  // ---- store (C layout: VGPR r -> row r + 8*lh, lane lr -> col) -----------
#pragma unroll
  for (int i = 0; i < 4; ++i)
#pragma unroll
    for (int j = 0; j < 2; ++j)
#pragma unroll
      for (int r = 0; r < 8; ++r)
        C[(size_t)(m0 + wm + i * 16 + r + 8 * lh) * N + n0 + wn + j * 16 + lr] =
            acc[i][j][r];
}

// ---------------------------------------------------------------------------
// Kernel 2: per-head RMSNorm (+weight) + RoPE, pack to f16.
//   Q -> [b,h,t,d] (score scale folded in), K -> [b,kv,t,d], V -> [b,kv,d,t]
// ---------------------------------------------------------------------------
__global__ __launch_bounds__(128) void norm_rope_pack(
    const float* __restrict__ qkv, const float* __restrict__ qw,
    const float* __restrict__ kw, _Float16* __restrict__ Q,
    _Float16* __restrict__ Kb, _Float16* __restrict__ Vt) {
  const int d = threadIdx.x;
  int idx     = blockIdx.x;
  const int s = idx % SLOTS_;  idx /= SLOTS_;
  const int t = idx % T_;
  const int b = idx / T_;

  const float v = qkv[((size_t)(b * T_ + t)) * QKVO_ + (size_t)s * D_ + d];

  __shared__ float sh[D_];
  __shared__ float red[4];

  if (s < H_ + KV_) {  // q or k: rmsnorm + rope
    float ss = v * v;
#pragma unroll
    for (int m = 1; m < 32; m <<= 1) ss += __shfl_xor(ss, m);
    if ((d & 31) == 0) red[d >> 5] = ss;
    __syncthreads();
    const float tot = red[0] + red[1] + red[2] + red[3];
    const float rms = rsqrtf(tot * (1.0f / D_) + 1e-6f);
    const float w   = (s < H_) ? qw[s * D_ + d] : kw[(s - H_) * D_ + d];
    sh[d] = v * rms * w;
    __syncthreads();

    const int   j   = d & 63;
    const float inv = __expf(-(float)(2 * j) * (1.0f / D_) * LN10K_);
    const float th  = (float)t * inv;
    const float c   = __cosf(th);
    const float sn  = __sinf(th);
    const float x1  = sh[j];
    const float x2  = sh[j + 64];
    float o = (d < 64) ? (x1 * c - x2 * sn) : (x1 * sn + x2 * c);

    if (s < H_) {
      o *= SCALE_;
      Q[(((size_t)b * H_ + s) * T_ + t) * D_ + d] = (_Float16)o;
    } else {
      Kb[(((size_t)b * KV_ + (s - H_)) * T_ + t) * D_ + d] = (_Float16)o;
    }
  } else {  // v: plain cast, transposed store
    const int kv = s - (H_ + KV_);
    Vt[(((size_t)b * KV_ + kv) * D_ + d) * T_ + t] = (_Float16)v;
  }
}

// ---------------------------------------------------------------------------
// Kernel 3: flash attention, one wave per 16 query rows, full 2048 context.
// Per 32-key step: 8 WMMA (scores) + online softmax in native C-tile layout
// + LDS transpose of P (C-layout -> A-layout) + 8 WMMA (PV).
// ---------------------------------------------------------------------------
__global__ __launch_bounds__(32) void attn_kernel(const _Float16* __restrict__ Q,
                                                  const _Float16* __restrict__ Kb,
                                                  const _Float16* __restrict__ Vt,
                                                  _Float16* __restrict__ AO) {
  __shared__ __align__(64) _Float16 Pl[16 * 32];

  int id       = blockIdx.x;
  const int qt = id & (T_ / 16 - 1);  id >>= 7;
  const int h  = id & (H_ - 1);
  const int b  = id >> 5;
  const int kv = h >> 3;  // GROUP = 8

  const _Float16* Qp = Q  + (((size_t)b * H_ + h) * T_ + (size_t)qt * 16) * D_;
  const _Float16* Kp = Kb + ((size_t)b * KV_ + kv) * (size_t)T_ * D_;
  const _Float16* Vp = Vt + ((size_t)b * KV_ + kv) * (size_t)D_ * T_;

  const int lane = threadIdx.x;
  const int lr   = lane & 15;
  const int lh   = lane >> 4;

  v16h qf[4];
#pragma unroll
  for (int p = 0; p < 4; ++p)
    qf[p] = a_frag(Qp + (size_t)lr * D_ + p * 32, lh);

  v8f   o[8] = {};
  float m[8], l[8];
#pragma unroll
  for (int r = 0; r < 8; ++r) { m[r] = -1e30f; l[r] = 0.0f; }

  for (int kt = 0; kt < T_; kt += 32) {
    // ---- scores: two 16x16 tiles over this 32-key block -------------------
    v8f s0 = {}, s1 = {};
#pragma unroll
    for (int p = 0; p < 4; ++p) {
      v16h k0 = *(const v16h*)(Kp + (size_t)(kt + lr) * D_ + p * 32 + lh * 16);
      v16h k1 = *(const v16h*)(Kp + (size_t)(kt + 16 + lr) * D_ + p * 32 + lh * 16);
      s0 = wmma_f16(qf[p], k0, s0);
      s1 = wmma_f16(qf[p], k1, s1);
    }

    // ---- online softmax in C-tile layout (row = r + 8*lh) -----------------
#pragma unroll
    for (int r = 0; r < 8; ++r) {
      float mx = fmaxf(s0[r], s1[r]);
      mx = fmaxf(mx, __shfl_xor(mx, 1));
      mx = fmaxf(mx, __shfl_xor(mx, 2));
      mx = fmaxf(mx, __shfl_xor(mx, 4));
      mx = fmaxf(mx, __shfl_xor(mx, 8));
      const float mn = fmaxf(m[r], mx);
      const float sc = __expf(m[r] - mn);
      m[r] = mn;
      l[r] *= sc;
#pragma unroll
      for (int t8 = 0; t8 < 8; ++t8) o[t8][r] *= sc;
      const float p0 = __expf(s0[r] - mn);
      const float p1 = __expf(s1[r] - mn);
      float rs = p0 + p1;
      rs += __shfl_xor(rs, 1);
      rs += __shfl_xor(rs, 2);
      rs += __shfl_xor(rs, 4);
      rs += __shfl_xor(rs, 8);
      l[r] += rs;
      Pl[(r + 8 * lh) * 32 + lr]      = (_Float16)p0;   // C-layout -> LDS
      Pl[(r + 8 * lh) * 32 + 16 + lr] = (_Float16)p1;
    }
    __syncthreads();

    // ---- PV: P (A-layout via LDS) x V (contiguous from transposed Vt) -----
    const v16h pa = a_frag(&Pl[lr * 32], lh);
#pragma unroll
    for (int t8 = 0; t8 < 8; ++t8) {
      v16h vf = *(const v16h*)(Vp + (size_t)(t8 * 16 + lr) * T_ + kt + lh * 16);
      o[t8] = wmma_f16(pa, vf, o[t8]);
    }
    __syncthreads();
  }

  // ---- normalize + store to AO[b*T+t][h*128+d] (f16, GEMM-ready) ----------
#pragma unroll
  for (int t8 = 0; t8 < 8; ++t8)
#pragma unroll
    for (int r = 0; r < 8; ++r) {
      const size_t row = (size_t)b * T_ + (size_t)qt * 16 + r + 8 * lh;
      AO[row * (size_t)(H_ * D_) + h * D_ + t8 * 16 + lr] =
          (_Float16)(o[t8][r] / l[r]);
    }
}

// ---------------------------------------------------------------------------
// Launch
// ---------------------------------------------------------------------------
extern "C" void kernel_launch(void* const* d_in, const int* in_sizes, int n_in,
                              void* d_out, int out_size, void* d_ws,
                              size_t ws_size, hipStream_t stream) {
  const float* x    = (const float*)d_in[0];  // [B,T,HID]
  const float* Wqkv = (const float*)d_in[1];  // [5120,4096]
  const float* Wo   = (const float*)d_in[2];  // [4096,4096]
  const float* qw   = (const float*)d_in[3];  // [32,128]
  const float* kw   = (const float*)d_in[4];  // [4,128]
  float* out        = (float*)d_out;          // [B,T,HID] fp32

  char* p = (char*)d_ws;
  float* qkv = (float*)p;
  p += (size_t)M_ * QKVO_ * sizeof(float);              // 84.0 MB
  _Float16* Qb = (_Float16*)p;
  p += (size_t)B_ * H_ * T_ * D_ * sizeof(_Float16);    // 33.6 MB
  _Float16* Kb = (_Float16*)p;
  p += (size_t)B_ * KV_ * T_ * D_ * sizeof(_Float16);   // 4.2 MB
  _Float16* Vt = (_Float16*)p;
  p += (size_t)B_ * KV_ * T_ * D_ * sizeof(_Float16);   // 4.2 MB
  _Float16* AO = (_Float16*)p;                          // 33.6 MB

  // 1) qkv = x @ Wqkv^T                       (fp32 -> f16 WMMA -> fp32)
  gemm_kernel<false><<<dim3(QKVO_ / 128, M_ / 128), 256, 0, stream>>>(
      x, Wqkv, qkv, M_, QKVO_, HID_);

  // 2) rmsnorm + rope + pack (Q scaled, V transposed)
  norm_rope_pack<<<B_ * T_ * SLOTS_, 128, 0, stream>>>(qkv, qw, kw, Qb, Kb, Vt);

  // 3) flash attention, one wave per 16 query rows
  attn_kernel<<<B_ * H_ * (T_ / 16), 32, 0, stream>>>(Qb, Kb, Vt, AO);

  // 4) out = AO @ Wo^T (A staged by Tensor Data Mover) (f16 WMMA -> fp32)
  gemm_kernel<true><<<dim3(HID_ / 128, M_ / 128), 256, 0, stream>>>(
      AO, Wo, out, M_, HID_, HID_);
}
